// ObstacleSelector_19533511262822
// MI455X (gfx1250) — matrix operand order repaired and checked by reading, not compiled, not run on hardware
//
#include <hip/hip_runtime.h>

typedef __attribute__((ext_vector_type(16))) _Float16 v16h;
typedef __attribute__((ext_vector_type(8)))  float    v8f;

static constexpr int BB = 16;        // batch
static constexpr int NN = 16;        // masks per batch
static constexpr int HH = 512;
static constexpr int WW = 512;
static constexpr int HW = HH * WW;   // 262144

// ---------------------------------------------------------------------------
// Kernel 0: zero the output (overlap slots are accumulated with atomics).
// ---------------------------------------------------------------------------
__global__ void zero_out_kernel(float* __restrict__ out, int n) {
    int i = blockIdx.x * blockDim.x + threadIdx.x;
    if (i < n) out[i] = 0.0f;
}

// ---------------------------------------------------------------------------
// Kernel 1: overlap[b,n] = sum_k masks[b,n,k] * target[b,k] via WMMA.
// One wave = one 16x16x32 f16 WMMA per k-chunk; A = 16 mask rows x 32 k,
// B = target chunk broadcast to all 16 columns (binary data -> f16 exact).
// Accumulator chained across chunks; integer-valued partials -> exact atomics.
// ---------------------------------------------------------------------------
__device__ __forceinline__ void cvt4(v16h& dst, int base, float4 v) {
    dst[base + 0] = (_Float16)v.x;
    dst[base + 1] = (_Float16)v.y;
    dst[base + 2] = (_Float16)v.z;
    dst[base + 3] = (_Float16)v.w;
}

__global__ __launch_bounds__(256)
void overlap_wmma_kernel(const float* __restrict__ masks,
                         const float* __restrict__ target,
                         float* __restrict__ out) {
    const int b          = blockIdx.y;
    const int lane       = threadIdx.x & 31;
    const int waveInBlk  = threadIdx.x >> 5;
    const int wavesPerB  = gridDim.x * (blockDim.x >> 5);
    const int waveId     = blockIdx.x * (blockDim.x >> 5) + waveInBlk;

    const int m  = lane & 15;   // mask row this lane feeds into A
    const int hi = lane >> 4;   // half-wave selector for k-sublayout

    const float* __restrict__ mrow = masks  + (size_t)(b * NN + m) * HW;
    const float* __restrict__ trow = target + (size_t)b * HW;

    v8f acc = {};

    for (int kc = waveId * 32; kc < HW; kc += wavesPerB * 32) {
        // ---- A: 16x32 f16. lane m: K 0-7 (v0-3) + 16-23 (v4-7);
        //                    lane m+16: K 8-15 + 24-31.
        const float4* pa0 = (const float4*)(mrow + kc +      hi * 8);
        const float4* pa1 = (const float4*)(mrow + kc + 16 + hi * 8);
        v16h a;
        cvt4(a, 0,  pa0[0]);
        cvt4(a, 4,  pa0[1]);
        cvt4(a, 8,  pa1[0]);
        cvt4(a, 12, pa1[1]);

        // ---- B: 32x16 f16, every column = target chunk.
        // lanes 0-15 hold K 0-7 + 16-23, lanes 16-31 hold K 8-15 + 24-31.
        const float4* pb0 = (const float4*)(trow + kc +      hi * 8);
        const float4* pb1 = (const float4*)(trow + kc + 16 + hi * 8);
        v16h bv;
        cvt4(bv, 0,  pb0[0]);
        cvt4(bv, 4,  pb0[1]);
        cvt4(bv, 8,  pb1[0]);
        cvt4(bv, 12, pb1[1]);

        acc = __builtin_amdgcn_wmma_f32_16x16x32_f16(
            /*neg_a=*/false, a, /*neg_b=*/false, bv,
            /*c_mod=*/(short)0, acc, /*reuse_a=*/false, /*reuse_b=*/false);
    }

    // D layout: VGPR r, lane j(0-15) -> D[M=r, N=j]; lane j+16 -> D[M=r+8, N=j].
    // All columns identical; lane 0 emits rows 0-7, lane 16 emits rows 8-15.
    if (lane == 0 || lane == 16) {
        const int rbase = (lane == 0) ? 0 : 8;
#pragma unroll
        for (int r = 0; r < 8; ++r) {
            atomicAdd(&out[(size_t)(b * NN + rbase + r) * 2 + 0], acc[r]);
        }
    }
}

// ---------------------------------------------------------------------------
// Kernel 2: target_area[b] = sum(target[b])  -> d_ws[b]
// ---------------------------------------------------------------------------
__global__ __launch_bounds__(256)
void area_kernel(const float* __restrict__ target, float* __restrict__ ws) {
    const int b = blockIdx.x;
    const float* __restrict__ t = target + (size_t)b * HW;

    float s = 0.0f;
    for (int i = threadIdx.x; i < HW; i += blockDim.x) s += t[i];

#pragma unroll
    for (int off = 16; off > 0; off >>= 1) s += __shfl_down(s, off, 32);

    __shared__ float sm[8];
    if ((threadIdx.x & 31) == 0) sm[threadIdx.x >> 5] = s;
    __syncthreads();
    if (threadIdx.x == 0) {
        float tot = 0.0f;
        for (int w = 0; w < (int)(blockDim.x >> 5); ++w) tot += sm[w];
        ws[b] = tot;
    }
}

// ---------------------------------------------------------------------------
// Kernel 3: iou[b,n] from box-windowed sum over target[b] (L2-resident).
// ---------------------------------------------------------------------------
__global__ __launch_bounds__(256)
void iou_kernel(const float* __restrict__ target,
                const int*   __restrict__ boxes,
                const float* __restrict__ ws,
                float* __restrict__ out) {
    const int bn = blockIdx.x;        // b*16 + n
    const int b  = bn >> 4;

    const int4 box = ((const int4*)boxes)[bn];  // x1,y1,x2,y2
    const int x1 = box.x, y1 = box.y, x2 = box.z, y2 = box.w;
    const int bw = x2 - x1;
    const int bh = y2 - y1;
    const int total = bw * bh;

    const float* __restrict__ t = target + (size_t)b * HW;

    float s = 0.0f;
    for (int i = threadIdx.x; i < total; i += blockDim.x) {
        const int r = i / bw;
        const int c = i - r * bw;
        s += t[(y1 + r) * WW + (x1 + c)];
    }

#pragma unroll
    for (int off = 16; off > 0; off >>= 1) s += __shfl_down(s, off, 32);

    __shared__ float sm[8];
    if ((threadIdx.x & 31) == 0) sm[threadIdx.x >> 5] = s;
    __syncthreads();
    if (threadIdx.x == 0) {
        float inter = 0.0f;
        for (int w = 0; w < (int)(blockDim.x >> 5); ++w) inter += sm[w];
        const float box_area = (float)(bw * bh);
        const float uni = box_area + ws[b] - inter;
        out[(size_t)bn * 2 + 1] = inter / (uni + 1e-8f);
    }
}

// ---------------------------------------------------------------------------
extern "C" void kernel_launch(void* const* d_in, const int* in_sizes, int n_in,
                              void* d_out, int out_size, void* d_ws, size_t ws_size,
                              hipStream_t stream) {
    const float* masks  = (const float*)d_in[0];  // (16,16,512,512) f32
    const float* target = (const float*)d_in[1];  // (16,512,512)    f32
    const int*   boxes  = (const int*)d_in[2];    // (16,16,4)       i32
    float* out = (float*)d_out;                   // (16,16,2)       f32
    float* ws  = (float*)d_ws;                    // 16 floats: target_area[b]

    // 1) zero overlap accumulators
    zero_out_kernel<<<dim3(2), dim3(256), 0, stream>>>(out, BB * NN * 2);

    // 2) target areas (one block per b, no atomics)
    area_kernel<<<dim3(BB), dim3(256), 0, stream>>>(target, ws);

    // 3) overlap einsum via WMMA: 64 slices x 16 batches, 8 waves/block
    overlap_wmma_kernel<<<dim3(64, BB), dim3(256), 0, stream>>>(masks, target, out);

    // 4) IoU: one block per (b,n)
    iou_kernel<<<dim3(BB * NN), dim3(256), 0, stream>>>(target, boxes, ws, out);
}
